// MultiHeadAttention_85375359910319
// MI455X (gfx1250) — matrix operand order repaired and checked by reading, not compiled, not run on hardware
//
#include <hip/hip_runtime.h>

typedef __attribute__((ext_vector_type(16))) __bf16 v16bf;
typedef __attribute__((ext_vector_type(8)))  float  v8f;
typedef uint32_t u32x4 __attribute__((ext_vector_type(4)));
typedef uint32_t u32x8 __attribute__((ext_vector_type(8)));

#define DMODEL 768
#define NHEAD  12
#define DK     64
#define SEQ    2048
#define BATCH  4

// ---------------------------------------------------------------------------
// Fragment loaders following CDNA5 WMMA VGPR striping (CDNA5 ISA):
//  A (16x32 bf16): lanes 0-15 row M=lane, K = k0..k0+7 then k0+16..k0+23
//                  lanes 16-31 same rows, k0 += 8.
//  B (32x16 bf16): lanes 0-15 col N=lane, K = 0..7 then 16..23; lanes 16-31 k += 8.
//  C (16x16 f32):  lanes 0-15 N=lane, M = vgpr (+8 for lanes 16-31).
// ---------------------------------------------------------------------------
__device__ inline v16bf frag_a(const __bf16* base, int ld, int koff) {
  int lane = threadIdx.x & 31;
  int row  = lane & 15;
  int k0   = koff + ((lane >> 4) << 3);
  const __bf16* p = base + (size_t)row * ld + k0;
  v16bf a;
  ((uint4*)&a)[0] = *(const uint4*)p;          // K = k0 .. k0+7
  ((uint4*)&a)[1] = *(const uint4*)(p + 16);   // K = k0+16 .. k0+23
  return a;
}

__device__ inline v16bf frag_b(const __bf16* base, int ld, int koff) {
  int lane = threadIdx.x & 31;
  int col  = lane & 15;
  int k0   = koff + ((lane >> 4) << 3);
  const __bf16* p = base + (size_t)col * ld + k0;
  v16bf b;
  ((uint4*)&b)[0] = *(const uint4*)p;          // K = k0 .. k0+7
  ((uint4*)&b)[1] = *(const uint4*)(p + 16);   // K = k0+16 .. k0+23
  return b;
}

// ---------------------------------------------------------------------------
// GEMM with bias: out[m,n] = sum_k X[m,k]*W[n,k] + bias[n]  (nn.Linear)
// X is fp32 or bf16 (converted to bf16 while staging into LDS); W is fp32.
// 64x64 block tile, K-step 32, 8 waves x 2 C-tiles.
// ---------------------------------------------------------------------------
template <typename TIN, typename TOUT>
__global__ __launch_bounds__(256) void gemm_bias_kernel(
    const TIN* __restrict__ X, const float* __restrict__ W,
    const float* __restrict__ bias, TOUT* __restrict__ out,
    int M, int N, int K) {
  __shared__ __align__(16) __bf16 As[64 * 32];
  __shared__ __align__(16) __bf16 Bs[64 * 32];

  const int tid = threadIdx.x;
  const int m0  = blockIdx.y * 64;
  const int n0  = blockIdx.x * 64;
  const int w   = tid >> 5;
  const int m16 = (w >> 1) << 4;
  const int nA  = (w & 1) << 5;

  const int sr = tid >> 2;      // staging row 0..63
  const int sq = tid & 3;       // staging quarter (8 elems each)

  v8f acc0 = {}, acc1 = {};

  for (int kk = 0; kk < K; kk += 32) {
    // Stage A tile (64 rows x 32 k), converting to bf16 if needed
    {
      const TIN* src = X + (size_t)(m0 + sr) * K + kk + sq * 8;
      __bf16* dst = As + sr * 32 + sq * 8;
      if constexpr (sizeof(TIN) == 4) {
        float4 f0 = ((const float4*)src)[0];
        float4 f1 = ((const float4*)src)[1];
        dst[0] = (__bf16)f0.x; dst[1] = (__bf16)f0.y;
        dst[2] = (__bf16)f0.z; dst[3] = (__bf16)f0.w;
        dst[4] = (__bf16)f1.x; dst[5] = (__bf16)f1.y;
        dst[6] = (__bf16)f1.z; dst[7] = (__bf16)f1.w;
      } else {
        *(uint4*)dst = *(const uint4*)src;
      }
    }
    // Stage B tile: rows of W are output columns (contraction along W cols)
    {
      const float* src = W + (size_t)(n0 + sr) * K + kk + sq * 8;
      __bf16* dst = Bs + sr * 32 + sq * 8;
      float4 f0 = ((const float4*)src)[0];
      float4 f1 = ((const float4*)src)[1];
      dst[0] = (__bf16)f0.x; dst[1] = (__bf16)f0.y;
      dst[2] = (__bf16)f0.z; dst[3] = (__bf16)f0.w;
      dst[4] = (__bf16)f1.x; dst[5] = (__bf16)f1.y;
      dst[6] = (__bf16)f1.z; dst[7] = (__bf16)f1.w;
    }
    __syncthreads();

    v16bf a  = frag_a(As + m16 * 32, 32, 0);
    v16bf b0 = frag_b(Bs + nA * 32, 32, 0);
    v16bf b1 = frag_b(Bs + (nA + 16) * 32, 32, 0);
    acc0 = __builtin_amdgcn_wmma_f32_16x16x32_bf16(false, a, false, b0,
                                                   (short)0, acc0, false, false);
    acc1 = __builtin_amdgcn_wmma_f32_16x16x32_bf16(false, a, false, b1,
                                                   (short)0, acc1, false, false);
    __syncthreads();
  }

  const int lane = tid & 31;
  const int col  = lane & 15;
  const int row0 = (lane >> 4) << 3;
#pragma unroll
  for (int sub = 0; sub < 2; ++sub) {
    v8f& acc = sub ? acc1 : acc0;
    int gcol = n0 + nA + sub * 16 + col;
    float bv = bias[gcol];
#pragma unroll
    for (int i = 0; i < 8; ++i) {
      int grow = m0 + m16 + row0 + i;
      float val = acc[i] + bv;
      if constexpr (sizeof(TOUT) == 4)
        out[(size_t)grow * N + gcol] = val;
      else
        out[(size_t)grow * N + gcol] = (__bf16)val;
    }
  }
}

// ---------------------------------------------------------------------------
// Flash-style attention for one (batch, head, 64-query tile).
// Q fragments live in VGPRs for the whole pass; K tile staged by the Tensor
// Data Mover (tensor_load_to_lds, wave 0 issues, TENSORcnt-synchronized),
// V tile staged transposed by all threads; online softmax with running row
// max/sum; O accumulator in registers.
// scores = (Q Kt)/8 -> mask==0 ? -1e9 -> * alpha -> softmax -> @ V
// ---------------------------------------------------------------------------
__global__ __launch_bounds__(256) void attention_kernel(
    const __bf16* __restrict__ Qp, const __bf16* __restrict__ Kp,
    const __bf16* __restrict__ Vp, const int* __restrict__ mask,
    const float* __restrict__ alphas, __bf16* __restrict__ attn) {
  __shared__ __align__(16) __bf16 Kt[64 * 64];    // [key][d] (TDM-loaded)
  __shared__ __align__(16) __bf16 VtT[64 * 64];   // [d][key] (transposed)
  __shared__ __align__(16) float  sbuf[64 * 64];  // scores / probs (f32)
  __shared__ __align__(16) __bf16 pbf[64 * 64];   // probs (bf16) for WMMA
  __shared__ float rowm[64], rowl[64], rowc[64];

  const int tid = threadIdx.x;
  const int q0  = blockIdx.x * 64;
  const int h   = blockIdx.y;
  const int b   = blockIdx.z;

  if (tid < 64) { rowm[tid] = -3.0e38f; rowl[tid] = 0.0f; }

  const int w    = tid >> 5;
  const int m16  = (w >> 1) << 4;
  const int nA   = (w & 1) << 5;
  const int lane = tid & 31;
  const int col  = lane & 15;
  const int row0 = (lane >> 4) << 3;

  // ---- TDM descriptor group 1 (constant across k-tiles) ----
  // data_size=2B; tensor_dim0 = DMODEL (row length); tensor_dim1 = 64 rows;
  // tile = 64 x 64; tensor_dim0_stride = DMODEL. 2D tile -> groups 2/3 unused.
  u32x8 g1;
  g1[0] = 0x00010000u;                 // workgroup_mask=0 | data_size=1 (2B)
  g1[1] = (uint32_t)DMODEL << 16;      // tensor_dim0[15:0] at bits 63:48
  g1[2] = 64u << 16;                   // tensor_dim0[31:16]=0 | tensor_dim1[15:0]
  g1[3] = 64u << 16;                   // tensor_dim1[31:16]=0 | tile_dim0=64
  g1[4] = 64u;                         // tile_dim1=64 | tile_dim2=0
  g1[5] = (uint32_t)DMODEL;            // tensor_dim0_stride[31:0]
  g1[6] = 0u;                          // stride hi | dim1_stride lo (unused, 2D)
  g1[7] = 0u;
  const uint32_t lds_kt = (uint32_t)(uintptr_t)(&Kt[0]);

  // Q fragments pinned in registers for the whole key loop
  const __bf16* Qbase = Qp + ((size_t)(b * SEQ + q0 + m16)) * DMODEL + h * DK;
  v16bf qa0 = frag_a(Qbase, DMODEL, 0);
  v16bf qa1 = frag_a(Qbase, DMODEL, 32);
  v8f o0 = {}, o1 = {};

  const int sr = tid >> 2;  // staging row 0..63
  const int sq = tid & 3;   // staging quarter (16 elems each)
  __syncthreads();

  for (int k0 = 0; k0 < SEQ; k0 += 64) {
    // Stage K tile via Tensor Data Mover: 64 rows x 128B, row stride 1536B,
    // packed contiguously into LDS -> natural [key][d] B-operand layout.
    if (tid < 32) {
      uint64_t ga = (uint64_t)(uintptr_t)(
          Kp + ((size_t)(b * SEQ + k0)) * DMODEL + h * DK);
      u32x4 g0;
      g0[0] = 1u;                                  // count=1, user descriptor
      g0[1] = lds_kt;                              // lds_addr
      g0[2] = (uint32_t)ga;                        // global_addr[31:0]
      g0[3] = (uint32_t)(ga >> 32) | (2u << 30);   // global_addr[56:32]|type=2
      asm volatile("tensor_load_to_lds %0, %1" :: "s"(g0), "s"(g1) : "memory");
    }
    // Stage V tile transposed -> [d][key] so PV B-fragments read contiguously
    {
      const __bf16* src =
          Vp + ((size_t)(b * SEQ + k0 + sr)) * DMODEL + h * DK + sq * 16;
#pragma unroll
      for (int i = 0; i < 16; ++i) VtT[(sq * 16 + i) * 64 + sr] = src[i];
    }
    // Prefetch next k-tile's mask/alpha rows (dominant HBM stream)
    if (k0 + 64 < SEQ) {
      size_t pidx = ((size_t)b * SEQ + (q0 + sr)) * SEQ + (k0 + 64) + sq * 16;
      __builtin_prefetch(&mask[pidx], 0, 1);
      __builtin_prefetch(&alphas[pidx], 0, 1);
    }
    __builtin_amdgcn_s_wait_tensorcnt(0);  // no-op for waves that didn't issue
    __syncthreads();

    // Raw scores: S = Q * K^T (contraction over d = 64, two K=32 WMMAs)
#pragma unroll
    for (int sub = 0; sub < 2; ++sub) {
      int n16 = nA + sub * 16;
      v8f c = {};
      c = __builtin_amdgcn_wmma_f32_16x16x32_bf16(
          false, qa0, false, frag_b(Kt + n16 * 64, 64, 0), (short)0, c, false, false);
      c = __builtin_amdgcn_wmma_f32_16x16x32_bf16(
          false, qa1, false, frag_b(Kt + n16 * 64, 64, 32), (short)0, c, false, false);
#pragma unroll
      for (int i = 0; i < 8; ++i)
        sbuf[(m16 + row0 + i) * 64 + n16 + col] = c[i];
    }
    __syncthreads();

    // scale 1/sqrt(64), mask, alpha (single pass over mask/alphas)
#pragma unroll
    for (int j = 0; j < 16; ++j) {
      int idx = tid * 16 + j;
      int rr = idx >> 6, cc = idx & 63;
      size_t gidx = ((size_t)b * SEQ + (q0 + rr)) * SEQ + (k0 + cc);
      float s = sbuf[idx] * 0.125f;
      s = (mask[gidx] == 0) ? -1.0e9f : s;
      s *= alphas[gidx];
      sbuf[idx] = s;
    }
    __syncthreads();

    // online-softmax: new row max + rescale factor
    if (tid < 64) {
      float tmax = -3.0e38f;
      for (int c2 = 0; c2 < 64; ++c2) tmax = fmaxf(tmax, sbuf[tid * 64 + c2]);
      float newm = fmaxf(rowm[tid], tmax);
      rowc[tid] = __expf(rowm[tid] - newm);
      rowm[tid] = newm;
    }
    __syncthreads();

    // exponentiate; write bf16 probs for WMMA
#pragma unroll
    for (int j = 0; j < 16; ++j) {
      int idx = tid * 16 + j;
      int rr = idx >> 6;
      float p = __expf(sbuf[idx] - rowm[rr]);
      sbuf[idx] = p;
      pbf[idx] = (__bf16)p;
    }
    __syncthreads();

    // running row sum (reads f32 probs)
    if (tid < 64) {
      float rs = 0.0f;
      for (int c2 = 0; c2 < 64; ++c2) rs += sbuf[tid * 64 + c2];
      rowl[tid] = rowl[tid] * rowc[tid] + rs;
    }

    // O = O * c + P @ V   (contraction over 64 keys, two K=32 WMMAs)
#pragma unroll
    for (int sub = 0; sub < 2; ++sub) {
      int d16 = nA + sub * 16;
      v8f pv = {};
      pv = __builtin_amdgcn_wmma_f32_16x16x32_bf16(
          false, frag_a(pbf + m16 * 64, 64, 0), false,
          frag_b(VtT + d16 * 64, 64, 0), (short)0, pv, false, false);
      pv = __builtin_amdgcn_wmma_f32_16x16x32_bf16(
          false, frag_a(pbf + m16 * 64, 64, 32), false,
          frag_b(VtT + d16 * 64, 64, 32), (short)0, pv, false, false);
      v8f& o = sub ? o1 : o0;
#pragma unroll
      for (int i = 0; i < 8; ++i) {
        float cfac = rowc[m16 + row0 + i];
        o[i] = o[i] * cfac + pv[i];
      }
    }
    __syncthreads();
  }

  // normalize and emit bf16 concat layout [B*S, D] (head h at cols h*64..)
#pragma unroll
  for (int sub = 0; sub < 2; ++sub) {
    int d16 = nA + sub * 16;
    v8f& o = sub ? o1 : o0;
#pragma unroll
    for (int i = 0; i < 8; ++i) {
      int grow = q0 + m16 + row0 + i;
      float denom = rowl[m16 + row0 + i];
      attn[((size_t)(b * SEQ + grow)) * DMODEL + h * DK + d16 + col] =
          (__bf16)(o[i] / denom);
    }
  }
}

// ---------------------------------------------------------------------------
extern "C" void kernel_launch(void* const* d_in, const int* in_sizes, int n_in,
                              void* d_out, int out_size, void* d_ws,
                              size_t ws_size, hipStream_t stream) {
  const float* q      = (const float*)d_in[0];
  const float* k      = (const float*)d_in[1];
  const float* v      = (const float*)d_in[2];
  const int*   mask   = (const int*)d_in[3];
  const float* alphas = (const float*)d_in[4];
  const float* Wq     = (const float*)d_in[5];
  const float* bq     = (const float*)d_in[6];
  const float* Wk     = (const float*)d_in[7];
  const float* bk     = (const float*)d_in[8];
  const float* Wv     = (const float*)d_in[9];
  const float* bv     = (const float*)d_in[10];
  const float* Wo     = (const float*)d_in[11];
  const float* bo     = (const float*)d_in[12];

  const size_t MT = (size_t)BATCH * SEQ;  // 8192 rows
  // workspace: 4 bf16 buffers of [8192 x 768] = ~50.3 MB total
  __bf16* Qp   = (__bf16*)d_ws;
  __bf16* Kp   = Qp + MT * DMODEL;
  __bf16* Vp   = Kp + MT * DMODEL;
  __bf16* attn = Vp + MT * DMODEL;

  dim3 gg(DMODEL / 64, (unsigned)(MT / 64));  // (12, 128) blocks
  gemm_bias_kernel<float, __bf16><<<gg, 256, 0, stream>>>(
      q, Wq, bq, Qp, (int)MT, DMODEL, DMODEL);
  gemm_bias_kernel<float, __bf16><<<gg, 256, 0, stream>>>(
      k, Wk, bk, Kp, (int)MT, DMODEL, DMODEL);
  gemm_bias_kernel<float, __bf16><<<gg, 256, 0, stream>>>(
      v, Wv, bv, Vp, (int)MT, DMODEL, DMODEL);

  dim3 ga(SEQ / 64, NHEAD, BATCH);  // (32, 12, 4)
  attention_kernel<<<ga, 256, 0, stream>>>(Qp, Kp, Vp, mask, alphas, attn);

  gemm_bias_kernel<__bf16, float><<<gg, 256, 0, stream>>>(
      attn, Wo, bo, (float*)d_out, (int)MT, DMODEL, DMODEL);
}
